// EncodeProcessDecode_44384192037530
// MI455X (gfx1250) — matrix operand order repaired and checked by reading, not compiled, not run on hardware
//
#include <hip/hip_runtime.h>
#include <hip/hip_bf16.h>
#include <stdint.h>

// ---------------------------------------------------------------------------
// EncodeProcessDecode GNN for MI455X (gfx1250), wave32 + WMMA bf16 + TDM.
// 32 rows per block (2 accumulators/wave) to halve L2 B-fragment traffic.
//
// Input flattening assumption (jax pytree: top-level insertion order, nested
// dicts sorted alphabetically, lists/tuples in order):
//   0: x [N,30] f32        1: edge_index [2,E] i32      2: edge_attr [E,4] f32
//   3..8:   dec   (W0,b0,W1,b1,W2,b2)          W:[in][out] row-major
//   9..16:  enc_edge (b, g, W0,b0,W1,b1,W2,b2)
//   17..24: enc_node (b, g, W0,b0,W1,b1,W2,b2)
//   25+16s: proc[s].edge (b,g,W0,b0,W1,b1,W2,b2), proc[s].node (same), s=0..9
//   185,186: unused scalars
// ---------------------------------------------------------------------------

typedef __attribute__((ext_vector_type(16))) __bf16         v16bf;
typedef __attribute__((ext_vector_type(8)))  float          v8f;
typedef __attribute__((ext_vector_type(4)))  unsigned short v4us;
typedef __attribute__((ext_vector_type(16))) unsigned short v16us;
typedef __attribute__((ext_vector_type(4)))  unsigned int   u32x4;
typedef __attribute__((ext_vector_type(8)))  int            i32x8;
typedef __attribute__((ext_vector_type(4)))  int            i32x4;

#define LAT 128
#define ROWS 32           // rows (M) per workgroup: 2 x 16-row WMMA tiles

// Tensor Data Mover availability (arity differs across toolchains)
#if __has_builtin(__builtin_amdgcn_tensor_load_to_lds)
#define HAVE_TDM 1
#else
#define HAVE_TDM 0
#endif

// RNE f32->bf16 (used once, in weight packing)
__device__ __forceinline__ unsigned short f2bf_rne(float f) {
  union { float f; unsigned int u; } cv; cv.f = f;
  unsigned int u = cv.u;
  unsigned int r = u + 0x7fffu + ((u >> 16) & 1u);
  return (unsigned short)(r >> 16);
}
// truncating f32->bf16 (activations; lowers to a d16_hi store / v_perm, ~free)
__device__ __forceinline__ unsigned short f2bf_t(float f) {
  union { float f; unsigned int u; } cv; cv.f = f;
  return (unsigned short)(cv.u >> 16);
}

// WMMA A-fragment LDS index inside one 16-row plane:
// Xs[(ks*32 + lane)*16 + h];  K = ks*32+kk:
//   lane = row + ((kk&8)?16:0) ;  h = (kk&7) + ((kk&16)?8:0)
__device__ __forceinline__ int afrag_idx(int r, int c) {
  const int ks = c >> 5, kk = c & 31;
  const int lane = r + ((kk & 8) ? 16 : 0);
  const int h = (kk & 7) + ((kk & 16) ? 8 : 0);
  return ((ks << 5) + lane) * 16 + h;
}

#if HAVE_TDM
// Build a tensor-DMA descriptor (D#) and issue TENSOR_LOAD_TO_LDS.
// w0: group0 word0 (count/gather bits). 2D tile or 2D gather (16-bit indices).
__device__ __forceinline__ void tdm_issue(unsigned w0, unsigned lds_off,
                                          const void* gp,
                                          int tdim0, int tdim1,
                                          int tildim0, int tildim1,
                                          unsigned stride0,
                                          i32x4 g2, i32x4 g3)
{
  const unsigned long long ga = (unsigned long long)gp;
  u32x4 g0;
  g0[0] = w0;                                   // count=1 [+ gather_mode]
  g0[1] = lds_off;                              // LDS byte address
  g0[2] = (unsigned)ga;                         // global_addr[31:0]
  g0[3] = (unsigned)((ga >> 32) & 0x01ffffffu) | (2u << 30);  // addr[56:32], type=2
  i32x8 g1;
  g1[0] = 0x20000;                                            // data_size=2 (4B)
  g1[1] = (int)(((unsigned)tdim0 & 0xffffu) << 16);           // tensor_dim0 lo
  g1[2] = (int)((((unsigned)tdim0 >> 16) & 0xffffu)
              | (((unsigned)tdim1 & 0xffffu) << 16));         // dim0 hi | dim1 lo
  g1[3] = (int)((((unsigned)tdim1 >> 16) & 0xffffu)
              | (((unsigned)tildim0 & 0xffffu) << 16));       // dim1 hi | tile_dim0
  g1[4] = (int)((unsigned)tildim1 & 0xffffu);                 // tile_dim1 (#rows/#idx)
  g1[5] = (int)stride0;                                       // tensor_dim0_stride lo
  g1[6] = 0;
  g1[7] = 0;
#if defined(__clang_major__) && __clang_major__ >= 23
  i32x8 gpad = {};
  __builtin_amdgcn_tensor_load_to_lds(g0, g1, g2, g3, gpad, 0);
#else
  __builtin_amdgcn_tensor_load_to_lds(g0, g1, g2, g3, 0);
#endif
}

__device__ __forceinline__ void tdm_wait0() {
#if __has_builtin(__builtin_amdgcn_s_wait_tensorcnt)
  __builtin_amdgcn_s_wait_tensorcnt(0);
#else
  asm volatile("s_wait_tensorcnt 0x0" ::: "memory");
#endif
}

// pack 16 int32 row indices into 8x16-bit pairs (gather descriptor groups 2/3)
__device__ __forceinline__ void pack_idx16(const int* p, i32x4& lo, i32x4& hi) {
#pragma unroll
  for (int j = 0; j < 4; ++j) {
    lo[j] = (p[2*j]     & 0xffff) | ((p[2*j + 1] & 0xffff) << 16);
    hi[j] = (p[2*j + 8] & 0xffff) | ((p[2*j + 9] & 0xffff) << 16);
  }
}
#endif // HAVE_TDM

struct Args {
  const float* x;         // [N,30]
  const float* edge_attr; // [E,4]
  const int*   src;       // [E]
  const int*   dst;       // [E]
  float* h;               // [N,128]
  float* e;               // [E,128]
  float* agg;             // [N,128]
  float* out;             // [N,3]
  const unsigned short* w1; const float* b1;
  const unsigned short* w2; const float* b2;
  const unsigned short* w3; const float* b3;
  const float* gam; const float* bet;     // LayerNorm gamma/beta
  int nrows;
  int nnodes;
};

// 32-row x 128-col GEMM layer. Wave wv owns column tile [16wv,16wv+16) and
// both 16-row planes: one B fragment load feeds two independent WMMAs.
// A planes and B are pre-packed in fragment order (contiguous 32B per lane).
// Epilogue handles each tile in its own loop so packed adds pair adjacent
// accumulator VGPRs (no cross-accumulator movs).
template<int KP2, bool OUTF32>
__device__ __forceinline__ void gemm_tile(const unsigned short* __restrict__ X,
                                          const unsigned short* __restrict__ Wp,
                                          const float* __restrict__ bias,
                                          unsigned short* obf, float* of32,
                                          bool relu, int ntiles, int ncols)
{
  const int lane = threadIdx.x & 31;
  const int wv   = threadIdx.x >> 5;
  if (wv >= ntiles) return;                 // wave-uniform: EXEC all-1s inside
  v8f acc0 = {}, acc1 = {};
#pragma unroll
  for (int ks = 0; ks < KP2 / 32; ++ks) {
    const int aoff = (((ks << 5) + lane) << 4);
    const v16us A0u = *(const v16us*)(X + aoff);                 // rows 0..15
    const v16us A1u = *(const v16us*)(X + 16 * KP2 + aoff);      // rows 16..31
    const v16us Bu  = *(const v16us*)(Wp + ((((ks * ntiles + wv) << 5) + lane) << 4));
    const v16bf A0 = __builtin_bit_cast(v16bf, A0u);
    const v16bf A1 = __builtin_bit_cast(v16bf, A1u);
    const v16bf B  = __builtin_bit_cast(v16bf, Bu);
    acc0 = __builtin_amdgcn_wmma_f32_16x16x32_bf16(false, A0, false, B, (short)0, acc0, false, false);
    acc1 = __builtin_amdgcn_wmma_f32_16x16x32_bf16(false, A1, false, B, (short)0, acc1, false, false);
  }
  const int rbase = (lane & 16) ? 8 : 0;    // C/D: VGPR g -> row g (+8 hi lanes)
  const int m     = lane & 15;
  const int col   = (wv << 4) | m;
  const float bb  = (col < ncols) ? bias[col] : 0.f;
#pragma unroll
  for (int g = 0; g < 8; ++g) {             // tile 0 (rows 0..15)
    float y = acc0[g] + bb;
    if (relu) y = fmaxf(y, 0.f);
    if (OUTF32) of32[(rbase + g) * LAT + col] = y;
    else        obf[afrag_idx(rbase + g, col)] = f2bf_t(y);
  }
#pragma unroll
  for (int g = 0; g < 8; ++g) {             // tile 1 (rows 16..31)
    float y = acc1[g] + bb;
    if (relu) y = fmaxf(y, 0.f);
    if (OUTF32) of32[(16 + rbase + g) * LAT + col] = y;
    else        obf[16 * LAT + afrag_idx(rbase + g, col)] = f2bf_t(y);
  }
}

// MODE: 0 enc_node(K=30->32)  1 enc_edge(K=4->32)  2 proc_edge(K=384)
//       3 proc_node(K=256)    4 decoder(K=128, out 3)
template<int KP, int MODE>
__global__ __launch_bounds__(256)
void mlp3_ln_kernel(Args a)
{
  // dynamic LDS: [ Yst | Xbf | midA | midB ]
  // Yst: LN buffer (ROWS*LAT f32). MODE2+TDM: 3-segment f32 staging area
  // (dst-gather / src-gather / e-rows); seg0 is reused as Yf after staging is
  // consumed; seg2 (e rows) stays live until the epilogue.
  extern __shared__ char smem_dyn[];
  constexpr int Y_FLOATS = (MODE == 2 && HAVE_TDM) ? 3 * ROWS * LAT : ROWS * LAT;
  float*          const Yst  = (float*)smem_dyn;
  unsigned short* const Xbf  = (unsigned short*)(smem_dyn + (size_t)Y_FLOATS * 4);
  unsigned short* const midA = Xbf + ROWS * KP;
  unsigned short* const midB = midA + ROWS * LAT;
  __shared__ float smu[ROWS], srs[ROWS];
  float* const Yf  = Yst;
  float* const stg = Yst;

  const int tid  = threadIdx.x;
  const int row0 = blockIdx.x * ROWS;

  // ---- stage (gather + concat + f32->bf16) into LDS in A-fragment order ----
  bool staged = false;
#if HAVE_TDM
  if (MODE == 2) {
    const unsigned ldsb = (unsigned)(size_t)(void*)stg;
    const unsigned GATHER = 1u | (1u << 31);   // count=1, gather_mode, 16b idx
    const unsigned PLAIN  = 1u;                // count=1
    const int SEG = ROWS * LAT * 4;            // 16KB per staging segment
    const int wvid = tid >> 5;
    if (wvid == 0) {                           // wave0: dst gathers + e tile
      const int* dp = a.dst + row0;
      i32x4 q2, q3, q2b, q3b;
      pack_idx16(dp,      q2,  q3);
      pack_idx16(dp + 16, q2b, q3b);
      const i32x4 z4 = {};
      tdm_issue(GATHER, ldsb,            a.h, 128, a.nnodes, 128, 16, 128, q2,  q3);
      tdm_issue(GATHER, ldsb + SEG / 2,  a.h, 128, a.nnodes, 128, 16, 128, q2b, q3b);
      tdm_issue(PLAIN,  ldsb + 2 * SEG,  a.e + (size_t)row0 * LAT,
                128, a.nrows - row0, 128, ROWS, 128, z4, z4);
      tdm_wait0();
    } else if (wvid == 1) {                    // wave1: src gathers
      const int* sp = a.src + row0;
      i32x4 q2, q3, q2b, q3b;
      pack_idx16(sp,      q2,  q3);
      pack_idx16(sp + 16, q2b, q3b);
      tdm_issue(GATHER, ldsb + SEG,           a.h, 128, a.nnodes, 128, 16, 128, q2,  q3);
      tdm_issue(GATHER, ldsb + SEG + SEG / 2, a.h, 128, a.nnodes, 128, 16, 128, q2b, q3b);
      tdm_wait0();
    }
    __syncthreads();                     // TDM data visible to all waves
    // convert f32 staging (LDS) -> bf16 A fragments
    for (int i = tid; i < ROWS * (KP / 4); i += 256) {
      const int r = i / (KP / 4), c4 = (i - r * (KP / 4)) * 4;
      const float4 v = *(const float4*)(stg + (c4 >> 7) * (ROWS * LAT)
                                            + r * LAT + (c4 & 127));
      v4us pk = { f2bf_t(v.x), f2bf_t(v.y), f2bf_t(v.z), f2bf_t(v.w) };
      *(v4us*)&Xbf[(r >> 4) * (16 * KP) + afrag_idx(r & 15, c4)] = pk;
    }
    staged = true;
  }
#endif
  if (!staged) {
    if (MODE == 0 || MODE == 1) {
      for (int i = tid; i < ROWS * KP; i += 256) {
        const int r = i / KP, c = i - r * KP;
        const int gr = row0 + r;
        float v = 0.f;
        if (gr < a.nrows) {
          if (MODE == 0) v = (c < 30) ? a.x[gr * 30 + c] : 0.f;
          else           v = (c < 4)  ? a.edge_attr[gr * 4 + c] : 0.f;
        }
        Xbf[(r >> 4) * (16 * KP) + afrag_idx(r & 15, c)] = f2bf_t(v);
      }
    } else {
      // all segments are 128-float aligned: 4 floats / thread (b128 -> b64)
      const int CPR = KP / 4;                       // chunks per row
      for (int i = tid; i < ROWS * CPR; i += 256) {
        const int r = i / CPR, c4 = (i - r * CPR) * 4;
        const int gr = row0 + r;
        float4 v = make_float4(0.f, 0.f, 0.f, 0.f);
        if (gr < a.nrows) {
          const float* sp;
          if (MODE == 2) {
            if (c4 < 128)      sp = a.h + (size_t)a.dst[gr] * LAT + c4;
            else if (c4 < 256) sp = a.h + (size_t)a.src[gr] * LAT + (c4 - 128);
            else               sp = a.e + (size_t)gr * LAT + (c4 - 256);
          } else if (MODE == 3) {
            sp = (c4 < 128) ? (a.agg + (size_t)gr * LAT + c4)
                            : (a.h   + (size_t)gr * LAT + (c4 - 128));
          } else {
            sp = a.h + (size_t)gr * LAT + c4;
          }
          v = *(const float4*)sp;
        }
        v4us pk = { f2bf_t(v.x), f2bf_t(v.y), f2bf_t(v.z), f2bf_t(v.w) };
        *(v4us*)&Xbf[(r >> 4) * (16 * KP) + afrag_idx(r & 15, c4)] = pk;
      }
    }
  }
  __syncthreads();

  gemm_tile<KP,  false>(Xbf,  a.w1, a.b1, midA, nullptr, true, 8, 128);
  __syncthreads();
  gemm_tile<LAT, false>(midA, a.w2, a.b2, midB, nullptr, true, 8, 128);
  __syncthreads();

  if (MODE == 4) {
    // final 128->3 (cols padded to one 16-wide tile; wave 0 only)
    gemm_tile<LAT, true>(midB, a.w3, a.b3, nullptr, Yf, false, 1, 3);
    __syncthreads();
    for (int i = tid; i < ROWS * 3; i += 256) {
      const int r = i / 3, c = i - r * 3;
      const int gr = row0 + r;
      if (gr < a.nrows) a.out[(size_t)gr * 3 + c] = Yf[r * LAT + c];
    }
    return;
  }

  gemm_tile<LAT, true>(midB, a.w3, a.b3, nullptr, Yf, false, 8, 128);
  __syncthreads();

  // ---- LayerNorm stats: 8 partials/row (16 elems each) + shuffle tree ----
  {
    const int r = tid >> 3, j = tid & 7;
    const float* yp = Yf + r * LAT + j * 16;
    float s = 0.f, s2 = 0.f;
#pragma unroll
    for (int q = 0; q < 4; ++q) {
      const float4 v = *(const float4*)(yp + 4 * q);
      s  += v.x + v.y + v.z + v.w;
      s2 += v.x*v.x + v.y*v.y + v.z*v.z + v.w*v.w;
    }
#pragma unroll
    for (int msk = 4; msk >= 1; msk >>= 1) {
      s  += __shfl_xor(s,  msk, 8);
      s2 += __shfl_xor(s2, msk, 8);
    }
    if (j == 0) {
      const float mu  = s * (1.f / LAT);
      const float var = s2 * (1.f / LAT) - mu * mu;
      smu[r] = mu;
      srs[r] = rsqrtf(var + 1e-5f);
    }
  }
  __syncthreads();

  // ---- normalize + affine + residual / scatter, 4 cols per thread ----
  for (int i = tid; i < ROWS * LAT / 4; i += 256) {
    const int r = i >> 5, c4 = (i & 31) * 4;
    const int gr = row0 + r;
    if (gr >= a.nrows) continue;
    const float mu = smu[r], rs = srs[r];
    float4 y = *(const float4*)(Yf + r * LAT + c4);
    const float4 gm = *(const float4*)(a.gam + c4);
    const float4 bt = *(const float4*)(a.bet + c4);
    float4 yn;
    yn.x = (y.x - mu) * rs * gm.x + bt.x;
    yn.y = (y.y - mu) * rs * gm.y + bt.y;
    yn.z = (y.z - mu) * rs * gm.z + bt.z;
    yn.w = (y.w - mu) * rs * gm.w + bt.w;
    if (MODE == 0) {
      *(float4*)(a.h + (size_t)gr * LAT + c4) = yn;
    } else if (MODE == 1) {
      *(float4*)(a.e + (size_t)gr * LAT + c4) = yn;
    } else if (MODE == 2) {
      // e += e_new  (old e rows still resident in LDS staging seg 2 w/ TDM)
#if HAVE_TDM
      float4 eo = *(const float4*)(stg + 2 * ROWS * LAT + r * LAT + c4);
#else
      float4 eo = *(const float4*)(a.e + (size_t)gr * LAT + c4);
#endif
      eo.x += yn.x; eo.y += yn.y; eo.z += yn.z; eo.w += yn.w;
      *(float4*)(a.e + (size_t)gr * LAT + c4) = eo;
      float* ap = a.agg + (size_t)a.dst[gr] * LAT + c4;    // segment_sum
      atomicAdd(ap + 0, yn.x);
      atomicAdd(ap + 1, yn.y);
      atomicAdd(ap + 2, yn.z);
      atomicAdd(ap + 3, yn.w);
    } else {
      float* hp = a.h + (size_t)gr * LAT + c4;             // h += h_new
      float4 ho = *(const float4*)hp;
      ho.x += yn.x; ho.y += yn.y; ho.z += yn.z; ho.w += yn.w;
      *(float4*)hp = ho;
    }
  }
}

// Pack f32 weight [K][N] (row-major in->out) into WMMA-B fragment order,
// padded: Kp = ceil32(K), cols padded to ntiles*16. Element
// out[((ks*ntiles+nt)*32 + lane)*16 + h] = W[ks*32 + (h&7) + (lane>=16?8:0) + (h>=8?16:0)][nt*16 + lane%16]
__global__ __launch_bounds__(256)
void pack_w_kernel(const float* __restrict__ W, int K, int N, int ntiles,
                   int ksteps, unsigned short* __restrict__ out)
{
  const int idx = blockIdx.x * 256 + threadIdx.x;
  const int total = ksteps * ntiles * 512;
  if (idx >= total) return;
  const int h    = idx & 15;
  const int lane = (idx >> 4) & 31;
  const int t    = idx >> 9;
  const int nt   = t % ntiles;
  const int ks   = t / ntiles;
  const int k    = ks * 32 + (h & 7) + ((lane & 16) ? 8 : 0) + ((h & 8) ? 16 : 0);
  const int n    = nt * 16 + (lane & 15);
  const float v  = (k < K && n < N) ? W[(size_t)k * N + n] : 0.f;
  out[idx] = f2bf_rne(v);
}

__global__ __launch_bounds__(256)
void zero_f32_kernel(float* p, int n)
{
  const int i = blockIdx.x * 256 + threadIdx.x;
  if (i < n) p[i] = 0.f;
}

// ---------------------------------------------------------------------------

extern "C" void kernel_launch(void* const* d_in, const int* in_sizes, int n_in,
                              void* d_out, int out_size, void* d_ws, size_t ws_size,
                              hipStream_t stream)
{
  const int N = in_sizes[0] / 30;
  const int E = in_sizes[2] / 4;

  const float* x         = (const float*)d_in[0];
  const int*   ei        = (const int*)  d_in[1];
  const float* edge_attr = (const float*)d_in[2];
  const int* src = ei;          // edge_index[0]
  const int* dst = ei + E;      // edge_index[1]

  int p = 3;
  const float *decW[3], *decB[3];
  for (int l = 0; l < 3; ++l) { decW[l] = (const float*)d_in[p++]; decB[l] = (const float*)d_in[p++]; }
  const float* eeBet = (const float*)d_in[p++];
  const float* eeGam = (const float*)d_in[p++];
  const float *eeW[3], *eeB[3];
  for (int l = 0; l < 3; ++l) { eeW[l] = (const float*)d_in[p++]; eeB[l] = (const float*)d_in[p++]; }
  const float* enBet = (const float*)d_in[p++];
  const float* enGam = (const float*)d_in[p++];
  const float *enW[3], *enB[3];
  for (int l = 0; l < 3; ++l) { enW[l] = (const float*)d_in[p++]; enB[l] = (const float*)d_in[p++]; }
  struct StepP { const float *bet, *gam, *W[3], *B[3]; };
  StepP pe[10], pn[10];
  for (int s = 0; s < 10; ++s) {
    pe[s].bet = (const float*)d_in[p++]; pe[s].gam = (const float*)d_in[p++];
    for (int l = 0; l < 3; ++l) { pe[s].W[l] = (const float*)d_in[p++]; pe[s].B[l] = (const float*)d_in[p++]; }
    pn[s].bet = (const float*)d_in[p++]; pn[s].gam = (const float*)d_in[p++];
    for (int l = 0; l < 3; ++l) { pn[s].W[l] = (const float*)d_in[p++]; pn[s].B[l] = (const float*)d_in[p++]; }
  }

  // ---- workspace carve (~188 MB: h 10.2 + agg 10.2 + e 163.8 + weights ~3.2)
  size_t cur = 0;
  auto carve = [&](size_t bytes) -> void* {
    void* r = (char*)d_ws + cur;
    cur += (bytes + 255) & ~(size_t)255;
    return r;
  };
  float* h   = (float*)carve((size_t)N * LAT * 4);
  float* agg = (float*)carve((size_t)N * LAT * 4);
  float* e   = (float*)carve((size_t)E * LAT * 4);

  auto pack = [&](const float* W, int K, int Ncols, int ntiles) -> unsigned short* {
    const int ksteps = (K + 31) / 32;
    const int total  = ksteps * ntiles * 512;
    unsigned short* dstp = (unsigned short*)carve((size_t)total * 2);
    pack_w_kernel<<<(total + 255) / 256, 256, 0, stream>>>(W, K, Ncols, ntiles, ksteps, dstp);
    return dstp;
  };

  unsigned short *enP[3], *eeP[3], *deP[3];
  unsigned short *peP[10][3], *pnP[10][3];
  enP[0] = pack(enW[0], 30, 128, 8);  enP[1] = pack(enW[1], 128, 128, 8);  enP[2] = pack(enW[2], 128, 128, 8);
  eeP[0] = pack(eeW[0], 4, 128, 8);   eeP[1] = pack(eeW[1], 128, 128, 8);  eeP[2] = pack(eeW[2], 128, 128, 8);
  for (int s = 0; s < 10; ++s) {
    peP[s][0] = pack(pe[s].W[0], 384, 128, 8);
    peP[s][1] = pack(pe[s].W[1], 128, 128, 8);
    peP[s][2] = pack(pe[s].W[2], 128, 128, 8);
    pnP[s][0] = pack(pn[s].W[0], 256, 128, 8);
    pnP[s][1] = pack(pn[s].W[1], 128, 128, 8);
    pnP[s][2] = pack(pn[s].W[2], 128, 128, 8);
  }
  deP[0] = pack(decW[0], 128, 128, 8);
  deP[1] = pack(decW[1], 128, 128, 8);
  deP[2] = pack(decW[2], 128, 3, 1);

  // dynamic LDS bytes per kernel instantiation: [Yst | Xbf | midA | midB]
  auto shbytes = [](int KP, int mode) -> size_t {
    const size_t y = (mode == 2 && HAVE_TDM) ? (size_t)3 * ROWS * LAT * 4
                                             : (size_t)ROWS * LAT * 4;
    return y + (size_t)ROWS * KP * 2 + 2 * (size_t)ROWS * LAT * 2;
  };

  const dim3 blk(256);
  const unsigned gN = (unsigned)((N + ROWS - 1) / ROWS);
  const unsigned gE = (unsigned)((E + ROWS - 1) / ROWS);

  Args a = {};
  a.x = x; a.edge_attr = edge_attr; a.src = src; a.dst = dst;
  a.h = h; a.e = e; a.agg = agg; a.out = (float*)d_out;
  a.nnodes = N;

  // encoder (node)
  a.w1 = enP[0]; a.b1 = enB[0]; a.w2 = enP[1]; a.b2 = enB[1]; a.w3 = enP[2]; a.b3 = enB[2];
  a.gam = enGam; a.bet = enBet; a.nrows = N;
  mlp3_ln_kernel<32, 0><<<gN, blk, shbytes(32, 0), stream>>>(a);

  // encoder (edge)
  a.w1 = eeP[0]; a.b1 = eeB[0]; a.w2 = eeP[1]; a.b2 = eeB[1]; a.w3 = eeP[2]; a.b3 = eeB[2];
  a.gam = eeGam; a.bet = eeBet; a.nrows = E;
  mlp3_ln_kernel<32, 1><<<gE, blk, shbytes(32, 1), stream>>>(a);

  // processor: 10 InteractionNetwork steps
  for (int s = 0; s < 10; ++s) {
    zero_f32_kernel<<<((unsigned)(N * LAT + 255)) / 256, blk, 0, stream>>>(agg, N * LAT);

    a.w1 = peP[s][0]; a.b1 = pe[s].B[0]; a.w2 = peP[s][1]; a.b2 = pe[s].B[1];
    a.w3 = peP[s][2]; a.b3 = pe[s].B[2]; a.gam = pe[s].gam; a.bet = pe[s].bet; a.nrows = E;
    mlp3_ln_kernel<384, 2><<<gE, blk, shbytes(384, 2), stream>>>(a);

    a.w1 = pnP[s][0]; a.b1 = pn[s].B[0]; a.w2 = pnP[s][1]; a.b2 = pn[s].B[1];
    a.w3 = pnP[s][2]; a.b3 = pn[s].B[2]; a.gam = pn[s].gam; a.bet = pn[s].bet; a.nrows = N;
    mlp3_ln_kernel<256, 3><<<gN, blk, shbytes(256, 3), stream>>>(a);
  }

  // decoder
  a.w1 = deP[0]; a.b1 = decB[0]; a.w2 = deP[1]; a.b2 = decB[1]; a.w3 = deP[2]; a.b3 = decB[2];
  a.nrows = N;
  mlp3_ln_kernel<128, 4><<<gN, blk, shbytes(128, 4), stream>>>(a);

  (void)n_in; (void)out_size; (void)ws_size;
}
